// MultivariateCausalFlow_67671504716231
// MI455X (gfx1250) — matrix operand order repaired and checked by reading, not compiled, not run on hardware
//
#include <hip/hip_runtime.h>
#include <hip/hip_bf16.h>
#include <math.h>

// ---------------------------------------------------------------------------
// Causal flow on gfx1250: per-wave 16-row tiles, v_wmma_f32_16x16x32_f16 GEMMs,
// activations resident in LDS across all 8 sequential steps, weights
// pre-converted to padded column-major f16 in workspace (L2-resident).
// ---------------------------------------------------------------------------

typedef __attribute__((ext_vector_type(16))) _Float16 v16h;
typedef __attribute__((ext_vector_type(8)))  _Float16 v8h;
typedef __attribute__((ext_vector_type(8)))  float    v8f;

#define DIMS    8
#define KDIM    64
#define TOTALF  512    // DIMS*KDIM
#define NH      100
#define NHP     112    // NH padded to 7 N-tiles of 16
#define KP      128    // hidden K padded to 4 K-tiles of 32
#define ROWS    16     // batch rows per wave (WMMA M)
#define WAVES   2      // waves per workgroup
#define TPB     (WAVES * 32)

// ---- A fragment (16x32 f16, row-major source, arbitrary row stride) --------
// lane&15 = row M; low half-wave holds K {k0..k0+15},
// high half-wave holds K {k0+16..k0+31}.
__device__ __forceinline__ v16h load_a_frag(const _Float16* row, int k0, int hi8) {
  return *(const v16h*)(row + k0 + 2 * hi8);
}

__device__ __forceinline__ float sigmoid_(float x) {
  return 1.0f / (1.0f + __expf(-x));
}

// ---- weight prep: f32 row-major [K x N] -> f16 column-major padded [Npad x Kpad]
__global__ void prep_w(const float* __restrict__ src, _Float16* __restrict__ dst,
                       int Kdim, int Ndim, int Kpad, int Npad) {
  int idx = blockIdx.x * blockDim.x + threadIdx.x;
  if (idx >= Kpad * Npad) return;
  int n = idx / Kpad;
  int k = idx - n * Kpad;
  float v = (k < Kdim && n < Ndim) ? src[(size_t)k * Ndim + n] : 0.0f;
  dst[(size_t)n * Kpad + k] = (_Float16)v;
}

__global__ void __launch_bounds__(TPB)
flow_kernel(const float* __restrict__ e, const int* __restrict__ C,
            const _Float16* __restrict__ sW1, const float* __restrict__ sb1,
            const _Float16* __restrict__ sW2, const float* __restrict__ sb2,
            const _Float16* __restrict__ sW3, const float* __restrict__ sb3,
            const _Float16* __restrict__ tW1, const float* __restrict__ tb1,
            const _Float16* __restrict__ tW2, const float* __restrict__ tb2,
            const _Float16* __restrict__ tW3, const float* __restrict__ tb3,
            float* __restrict__ zout, float* __restrict__ logdet) {
  // Per-wave LDS regions (no inter-wave sharing -> no barriers needed).
  __shared__ _Float16 Xs [WAVES][ROWS][TOTALF];  // z tile, f16, updated per step
  __shared__ _Float16 H1s[WAVES][ROWS][KP];      // hidden 1 (zero-padded K)
  __shared__ _Float16 H2s[WAVES][ROWS][KP];      // hidden 2
  __shared__ float    Sst[WAVES][ROWS][KDIM];    // sigmoid(s) staging

  const int tid     = threadIdx.x;
  const int wave    = tid >> 5;
  const int lane    = tid & 31;
  const int rowbase = blockIdx.x * (WAVES * ROWS) + wave * ROWS;

  _Float16 (*X )[TOTALF] = Xs [wave];
  _Float16 (*H1)[KP]     = H1s[wave];
  _Float16 (*H2)[KP]     = H2s[wave];
  float    (*Sb)[KDIM]   = Sst[wave];

  // zero-init z tile (z0 = 0) and hidden buffers (establishes zero K-padding)
  {
    unsigned int* px = (unsigned int*)&X[0][0];
    for (int q = lane; q < ROWS * TOTALF / 2; q += 32) px[q] = 0u;
    unsigned int* p1 = (unsigned int*)&H1[0][0];
    unsigned int* p2 = (unsigned int*)&H2[0][0];
    for (int q = lane; q < ROWS * KP / 2; q += 32) { p1[q] = 0u; p2[q] = 0u; }
  }
  asm volatile("s_wait_dscnt 0x0" ::: "memory");

  const int ncol = lane & 15;          // WMMA N / A-row index
  const int hi8  = (lane >> 4) << 3;   // half-wave row offset for C/D
  const int hi16 = (lane >> 4) << 4;   // half-wave K offset for B
  const _Float16* arowX  = &X [ncol][0];
  const _Float16* arowH1 = &H1[ncol][0];
  const _Float16* arowH2 = &H2[ncol][0];

  float ldacc = 0.0f;

  for (int i = 0; i < DIMS; ++i) {
    // mask column C[:, i]; generic: scale each 64-feature K-block by C[j][i]
    float csc[DIMS];
    for (int j = 0; j < DIMS; ++j) csc[j] = (float)C[j * DIMS + i];

    for (int pass = 0; pass < 2; ++pass) {  // pass 0 = s-net, pass 1 = t-net
      const _Float16* W1 = pass ? tW1 : sW1;
      const float*    b1 = pass ? tb1 : sb1;
      const _Float16* W2 = pass ? tW2 : sW2;
      const float*    b2 = pass ? tb2 : sb2;
      const _Float16* W3 = pass ? tW3 : sW3;
      const float*    b3 = pass ? tb3 : sb3;

      // ---------- layer 1: [16 x 512] @ [512 x 112] -> relu -> H1 ----------
      {
        v8f acc[7];
        #pragma unroll
        for (int nt = 0; nt < 7; ++nt) acc[nt] = (v8f){0,0,0,0,0,0,0,0};
        for (int j = 0; j < DIMS; ++j) {
          const float mj = csc[j];
          if (mj == 0.0f) continue;               // uniform branch: skip masked block
          const bool need_scale = (mj != 1.0f);
          const _Float16 mh = (_Float16)mj;
          #pragma unroll
          for (int kk = 0; kk < 2; ++kk) {
            const int k0 = j * 64 + kk * 32;
            v16h a = load_a_frag(arowX, k0, hi8);
            if (need_scale) {
              #pragma unroll
              for (int q = 0; q < 16; ++q) a[q] = (_Float16)(a[q] * mh);
            }
            #pragma unroll
            for (int nt = 0; nt < 7; ++nt) {
              const v16h b = *(const v16h*)(W1 + (size_t)(nt * 16 + ncol) * TOTALF + k0 + hi16);
              acc[nt] = __builtin_amdgcn_wmma_f32_16x16x32_f16(
                  false, a, false, b, (short)0, acc[nt], false, false);
            }
          }
        }
        #pragma unroll
        for (int nt = 0; nt < 7; ++nt) {
          const int n = nt * 16 + ncol;
          const float bv = (n < NH) ? b1[n] : 0.0f;
          #pragma unroll
          for (int r = 0; r < 8; ++r) {
            float h = acc[nt][r] + bv;
            h = h > 0.0f ? h : 0.0f;
            H1[r + hi8][n] = (_Float16)h;        // n in [100,112): bias 0, W cols 0 -> stays 0
          }
        }
      }
      asm volatile("s_wait_dscnt 0x0" ::: "memory");

      // ---------- layer 2: [16 x 128] @ [128 x 112] -> relu -> H2 ----------
      {
        v8f acc[7];
        #pragma unroll
        for (int nt = 0; nt < 7; ++nt) acc[nt] = (v8f){0,0,0,0,0,0,0,0};
        #pragma unroll
        for (int kt = 0; kt < 4; ++kt) {
          const int k0 = kt * 32;
          const v16h a = load_a_frag(arowH1, k0, hi8);
          #pragma unroll
          for (int nt = 0; nt < 7; ++nt) {
            const v16h b = *(const v16h*)(W2 + (size_t)(nt * 16 + ncol) * KP + k0 + hi16);
            acc[nt] = __builtin_amdgcn_wmma_f32_16x16x32_f16(
                false, a, false, b, (short)0, acc[nt], false, false);
          }
        }
        #pragma unroll
        for (int nt = 0; nt < 7; ++nt) {
          const int n = nt * 16 + ncol;
          const float bv = (n < NH) ? b2[n] : 0.0f;
          #pragma unroll
          for (int r = 0; r < 8; ++r) {
            float h = acc[nt][r] + bv;
            h = h > 0.0f ? h : 0.0f;
            H2[r + hi8][n] = (_Float16)h;
          }
        }
      }
      asm volatile("s_wait_dscnt 0x0" ::: "memory");

      // ---------- layer 3: [16 x 128] @ [128 x 64] -> sigmoid --------------
      {
        v8f acc[4];
        #pragma unroll
        for (int nt = 0; nt < 4; ++nt) acc[nt] = (v8f){0,0,0,0,0,0,0,0};
        #pragma unroll
        for (int kt = 0; kt < 4; ++kt) {
          const int k0 = kt * 32;
          const v16h a = load_a_frag(arowH2, k0, hi8);
          #pragma unroll
          for (int nt = 0; nt < 4; ++nt) {
            const v16h b = *(const v16h*)(W3 + (size_t)(nt * 16 + ncol) * KP + k0 + hi16);
            acc[nt] = __builtin_amdgcn_wmma_f32_16x16x32_f16(
                false, a, false, b, (short)0, acc[nt], false, false);
          }
        }
        if (pass == 0) {
          // stage sigmoid(s) for log_det and the z-update
          #pragma unroll
          for (int nt = 0; nt < 4; ++nt) {
            const int n = nt * 16 + ncol;
            const float bv = b3[n];
            #pragma unroll
            for (int r = 0; r < 8; ++r)
              Sb[r + hi8][n] = sigmoid_(acc[nt][r] + bv);
          }
        } else {
          // fused epilogue: z[:,i,:] = exp(s)*e_i + t ; write global + f16 tile
          #pragma unroll
          for (int nt = 0; nt < 4; ++nt) {
            const int n = nt * 16 + ncol;
            const float bv = b3[n];
            #pragma unroll
            for (int r = 0; r < 8; ++r) {
              const int m = r + hi8;
              const float tval = sigmoid_(acc[nt][r] + bv);
              const float sval = Sb[m][n];
              const size_t gidx = (size_t)(rowbase + m) * TOTALF + (size_t)i * KDIM + n;
              const float zv = __expf(sval) * e[gidx] + tval;
              zout[gidx] = zv;
              X[m][i * KDIM + n] = (_Float16)zv;
            }
          }
        }
      }
      asm volatile("s_wait_dscnt 0x0" ::: "memory");

      if (pass == 0) {
        // log_det partial: lane m (<16) sums its row of sigmoid(s)
        if (lane < 16) {
          const float* row = &Sb[lane][0];
          float ssum = 0.0f;
          #pragma unroll
          for (int c = 0; c < 16; ++c) {
            const float4 v = *(const float4*)(row + 4 * c);
            ssum += (v.x + v.y) + (v.z + v.w);
          }
          ldacc += ssum;
        }
      }
    } // pass
  } // step i

  if (lane < 16) logdet[rowbase + lane] = ldacc;
}

extern "C" void kernel_launch(void* const* d_in, const int* in_sizes, int n_in,
                              void* d_out, int out_size, void* d_ws, size_t ws_size,
                              hipStream_t stream) {
  const float* e   = (const float*)d_in[0];
  const int*   C   = (const int*)  d_in[1];
  const float* sW1 = (const float*)d_in[2];
  const float* sb1 = (const float*)d_in[3];
  const float* sW2 = (const float*)d_in[4];
  const float* sb2 = (const float*)d_in[5];
  const float* sW3 = (const float*)d_in[6];
  const float* sb3 = (const float*)d_in[7];
  const float* tW1 = (const float*)d_in[8];
  const float* tb1 = (const float*)d_in[9];
  const float* tW2 = (const float*)d_in[10];
  const float* tb2 = (const float*)d_in[11];
  const float* tW3 = (const float*)d_in[12];
  const float* tb3 = (const float*)d_in[13];

  const int Btot = in_sizes[0] / TOTALF;  // 65536

  // workspace layout: f16 column-major padded weights (~312 KB total)
  char* ws = (char*)d_ws;
  _Float16* sW1c = (_Float16*)(ws + 0);        // 112*512*2 = 114688
  _Float16* tW1c = (_Float16*)(ws + 114688);   // 114688
  _Float16* sW2c = (_Float16*)(ws + 229376);   // 112*128*2 = 28672
  _Float16* tW2c = (_Float16*)(ws + 258048);   // 28672
  _Float16* sW3c = (_Float16*)(ws + 286720);   //  64*128*2 = 16384
  _Float16* tW3c = (_Float16*)(ws + 303104);   // 16384  (end: 319488)

  const int PT = 256;
  prep_w<<<(NHP * TOTALF + PT - 1) / PT, PT, 0, stream>>>(sW1, sW1c, TOTALF, NH, TOTALF, NHP);
  prep_w<<<(NHP * TOTALF + PT - 1) / PT, PT, 0, stream>>>(tW1, tW1c, TOTALF, NH, TOTALF, NHP);
  prep_w<<<(NHP * KP     + PT - 1) / PT, PT, 0, stream>>>(sW2, sW2c, NH, NH, KP, NHP);
  prep_w<<<(NHP * KP     + PT - 1) / PT, PT, 0, stream>>>(tW2, tW2c, NH, NH, KP, NHP);
  prep_w<<<(KDIM * KP    + PT - 1) / PT, PT, 0, stream>>>(sW3, sW3c, NH, KDIM, KP, KDIM);
  prep_w<<<(KDIM * KP    + PT - 1) / PT, PT, 0, stream>>>(tW3, tW3c, NH, KDIM, KP, KDIM);

  float* zout = (float*)d_out;
  float* ld   = zout + (size_t)Btot * TOTALF;

  const int blocks = Btot / (WAVES * ROWS);  // 2048
  flow_kernel<<<blocks, TPB, 0, stream>>>(e, C,
                                          sW1c, sb1, sW2c, sb2, sW3c, sb3,
                                          tW1c, tb1, tW2c, tb2, tW3c, tb3,
                                          zout, ld);
}
